// NystromAttention2D_78134045048861
// MI455X (gfx1250) — matrix operand order repaired and checked by reading, not compiled, not run on hardware
//
#include <hip/hip_runtime.h>
#include <hip/hip_bf16.h>
#include <stdint.h>

typedef __attribute__((ext_vector_type(16))) __bf16 v16bf;
typedef __attribute__((ext_vector_type(8)))  float  v8f;
typedef __attribute__((ext_vector_type(4)))  unsigned int uint4v;
typedef __attribute__((ext_vector_type(8)))  int int8v;
typedef __attribute__((ext_vector_type(4)))  int int4v;

union Frag16 { v16bf v; unsigned short u[16]; };

static constexpr int N_SEQ  = 32768;
static constexpr int DIM    = 512;
static constexpr int H      = 8;
static constexpr int DH     = 64;
static constexpr int INNER  = 512;
static constexpr int LSEG   = 64;
static constexpr int KS     = 33;
static constexpr int THREE_INNER = 1536;
static constexpr int SORT_BLOCKS = N_SEQ / 256; // 128

// GEMM LDS tiling: padded row stride 40 u16 (80B, 16B aligned, set by TDM pad)
static constexpr int AROW_U16  = 40;
static constexpr int ATILE_U16 = 128 * AROW_U16;            // 5120 u16
static constexpr int BTILE_U16 = 64 * AROW_U16;             // 2560 u16
static constexpr int BUF_U16   = ATILE_U16 + BTILE_U16;     // 7680 u16
static constexpr unsigned GEMM_SMEM_BYTES = 2u * BUF_U16 * sizeof(unsigned short); // 30720

#if defined(__has_builtin)
#  if __has_builtin(__builtin_amdgcn_tensor_load_to_lds)
#    define HAS_TDM 1
#  endif
#endif
#ifndef HAS_TDM
#  define HAS_TDM 0
#endif

__device__ __forceinline__ unsigned short f2bf(float f) {
  unsigned int u = __builtin_bit_cast(unsigned int, f);
  u += 0x7FFFu + ((u >> 16) & 1u);           // round-to-nearest-even
  return (unsigned short)(u >> 16);
}

#if HAS_TDM
// Issue a 2D TDM tile load: tile (tile_d0 x tile_d1) of 2-byte elements from a
// row-major tensor (row length tensor_d0, stride stride_elems) into LDS at
// lds_off, inserting 4 DWORDs of padding after every 16 DWORDs (64B rows ->
// 80B LDS row stride). D# layout per CDNA5 ISA section 8.
// This toolchain's builtin takes 6 args (therock-10.0 style):
//   (uint32x4 g0, int32x8 g1, int32x4 g2, int32x4 g3, int32x8 extra, i32 cpol)
__device__ __forceinline__ void tdm_load_2d(unsigned lds_off, unsigned long long gaddr,
                                            int tensor_d0, int tensor_d1,
                                            int tile_d0, int tile_d1, int stride_elems) {
  uint4v g0;
  g0[0] = 1u;                                              // count=1, load, user mode
  g0[1] = lds_off;                                         // lds_addr (bytes)
  g0[2] = (unsigned)(gaddr & 0xFFFFFFFFull);               // global_addr[31:0]
  g0[3] = ((unsigned)(gaddr >> 32) & 0x1FFFFFFu) | (2u << 30); // addr[56:32] | type=2
  int8v g1;
  g1[0] = (1 << 16)      // data_size = 2 bytes
        | (1 << 20)      // pad_enable
        | (3 << 22)      // pad_interval: 16 DWORDs (64B)
        | (3 << 25);     // pad_amount: 4 DWORDs (16B)
  g1[1] = (tensor_d0 & 0xFFFF) << 16;                      // tensor_dim0[15:0]
  g1[2] = ((tensor_d0 >> 16) & 0xFFFF) | ((tensor_d1 & 0xFFFF) << 16);
  g1[3] = ((tensor_d1 >> 16) & 0xFFFF) | (tile_d0 << 16);  // tile_dim0
  g1[4] = tile_d1;                                         // tile_dim1 (tile_dim2=0)
  g1[5] = stride_elems;                                    // tensor_dim0_stride[31:0]
  g1[6] = 0;
  g1[7] = 0;
  int4v z4 = {0, 0, 0, 0};
  int8v z8 = {0, 0, 0, 0, 0, 0, 0, 0};
  __builtin_amdgcn_tensor_load_to_lds(g0, g1, z4, z4, z8, 0);
}
#endif

// ---------------------------------------------------------------------------
// Stable counting sort of labels (L=64): hist -> scan -> scatter
// ---------------------------------------------------------------------------
__global__ __launch_bounds__(256)
void hist_kernel(const int* __restrict__ labels, int* __restrict__ blockHist) {
  __shared__ int hist[64];
  const int t = threadIdx.x;
  if (t < 64) hist[t] = 0;
  __syncthreads();
  const int lbl = labels[blockIdx.x * 256 + t];
  atomicAdd(&hist[lbl], 1);
  __syncthreads();
  if (t < 64) blockHist[blockIdx.x * 64 + t] = hist[t];
}

__global__ void scan_kernel(const int* __restrict__ blockHist,
                            int* __restrict__ blockOff, int* __restrict__ base) {
  __shared__ int totals[64];
  __shared__ int baseS[64];
  const int l = threadIdx.x;                  // 64 threads
  int run = 0;
  for (int b = 0; b < SORT_BLOCKS; ++b) {
    blockOff[b * 64 + l] = run;
    run += blockHist[b * 64 + l];
  }
  totals[l] = run;
  __syncthreads();
  if (l == 0) {
    int acc = 0;
    for (int j = 0; j < 64; ++j) { baseS[j] = acc; acc += totals[j]; }
    base[64] = acc;                           // == N
  }
  __syncthreads();
  base[l] = baseS[l];
  for (int b = 0; b < SORT_BLOCKS; ++b) blockOff[b * 64 + l] += baseS[l];
}

__global__ __launch_bounds__(256)
void scatter_kernel(const int* __restrict__ labels,
                    const int* __restrict__ blockOff, int* __restrict__ perm) {
  __shared__ int lbl[256];
  const int t = threadIdx.x;
  const int i = blockIdx.x * 256 + t;
  lbl[t] = labels[i];
  __syncthreads();
  const int my = lbl[t];
  int rank = 0;
  for (int j = 0; j < t; ++j) rank += (lbl[j] == my) ? 1 : 0;
  perm[blockOff[blockIdx.x * 64 + my] + rank] = i;   // perm[sorted_pos] = orig idx
}

// ---------------------------------------------------------------------------
// Pre-pass conversions for the WMMA GEMMs
// ---------------------------------------------------------------------------
__global__ __launch_bounds__(256)
void gather_x_bf16(const float* __restrict__ x, const int* __restrict__ perm,
                   unsigned short* __restrict__ xbf) {
  const int t = blockIdx.x * 256 + threadIdx.x;     // N*DIM/4 threads
  const int p = t >> 7;                             // 128 float4 per row
  const int c4 = (t & 127) * 4;
  const float4 f = *reinterpret_cast<const float4*>(x + (size_t)perm[p] * DIM + c4);
  unsigned short* dst = xbf + (size_t)p * DIM + c4;
  dst[0] = f2bf(f.x); dst[1] = f2bf(f.y); dst[2] = f2bf(f.z); dst[3] = f2bf(f.w);
}

// Wt[n][k] = bf16(W[k][n])
__global__ __launch_bounds__(256)
void transpose_bf16(const float* __restrict__ W, unsigned short* __restrict__ Wt,
                    int K, int Ncols) {
  const int idx = blockIdx.x * 256 + threadIdx.x;
  if (idx >= K * Ncols) return;
  const int n = idx / K, k = idx - n * K;           // writes coalesced over k
  Wt[(size_t)n * K + k] = f2bf(W[(size_t)k * Ncols + n]);
}

// ---------------------------------------------------------------------------
// bf16 WMMA GEMM with TDM double-buffered LDS pipeline.
// A: [M=32768][Kdim] bf16 row-major.  Bt: [Ncols][Kdim] bf16 (K-major).
// Block tile 128x64x32, 8 waves, each wave 2x2 16x16 WMMA tiles.
// MODE 0: store into q/k/v [H][N][DH].  MODE 1: store out[perm[m]] + bias.
// ---------------------------------------------------------------------------
template <int MODE>
__global__ __launch_bounds__(256)
void gemm_bf16_wmma(const unsigned short* __restrict__ Abf,
                    const unsigned short* __restrict__ Btbf,
                    int Kdim, int Brows,
                    float* __restrict__ qo, float* __restrict__ ko, float* __restrict__ vo,
                    const int* __restrict__ permOut, const float* __restrict__ bias,
                    float* __restrict__ out) {
  extern __shared__ unsigned short smem[];
  const int t      = threadIdx.x;
  const int lane   = t & 31;
  const int wid    = t >> 5;
  const int waveM  = (wid & 3) * 32;
  const int waveN  = (wid >> 2) * 32;
  const int blockN = blockIdx.x * 64;
  const int blockM = blockIdx.y * 128;
  const int nK     = Kdim / 32;

  v8f acc[2][2] = {};

#if HAS_TDM
  if (wid == 0) {
    tdm_load_2d(0u, (unsigned long long)(uintptr_t)Abf + ((size_t)blockM * Kdim) * 2,
                Kdim, N_SEQ, 32, 128, Kdim);
    tdm_load_2d((unsigned)(ATILE_U16 * 2),
                (unsigned long long)(uintptr_t)Btbf + ((size_t)blockN * Kdim) * 2,
                Kdim, Brows, 32, 64, Kdim);
  }
#endif

  for (int kt = 0; kt < nK; ++kt) {
    const int buf = kt & 1;
    const unsigned short* sA = smem + buf * BUF_U16;
    const unsigned short* sB = sA + ATILE_U16;

#if HAS_TDM
    if (wid == 0) {
      if (kt + 1 < nK) {
        const unsigned ldsA = (unsigned)((buf ^ 1) * BUF_U16 * 2);
        tdm_load_2d(ldsA,
                    (unsigned long long)(uintptr_t)Abf + ((size_t)blockM * Kdim + (kt + 1) * 32) * 2,
                    Kdim, N_SEQ, 32, 128, Kdim);
        tdm_load_2d(ldsA + (unsigned)(ATILE_U16 * 2),
                    (unsigned long long)(uintptr_t)Btbf + ((size_t)blockN * Kdim + (kt + 1) * 32) * 2,
                    Kdim, Brows, 32, 64, Kdim);
        __builtin_amdgcn_s_wait_tensorcnt((short)2);   // drain current tile only
      } else {
        __builtin_amdgcn_s_wait_tensorcnt((short)0);
      }
    }
    __syncthreads();
#else
    {
      unsigned short* dA = smem + buf * BUF_U16;
      unsigned short* dB = dA + ATILE_U16;
      for (int e = t; e < 128 * 8; e += 256) {          // A: 128 rows x 32 u16
        const int row = e >> 3, g = e & 7;
        const uint2 d2 = *reinterpret_cast<const uint2*>(
            Abf + (size_t)(blockM + row) * Kdim + kt * 32 + g * 4);
        *reinterpret_cast<uint2*>(dA + row * AROW_U16 + g * 4) = d2;
      }
      for (int e = t; e < 64 * 8; e += 256) {           // B: 64 rows x 32 u16
        const int row = e >> 3, g = e & 7;
        const uint2 d2 = *reinterpret_cast<const uint2*>(
            Btbf + (size_t)(blockN + row) * Kdim + kt * 32 + g * 4);
        *reinterpret_cast<uint2*>(dB + row * AROW_U16 + g * 4) = d2;
      }
      __syncthreads();
    }
#endif

    // ---- fragments: contiguous k = 16*half + s for both A and B (wave32)
    const int hi  = lane >> 4;
    const int l15 = lane & 15;
    Frag16 fa[2], fb[2];
#pragma unroll
    for (int tm = 0; tm < 2; ++tm) {
      const unsigned short* rowp = sA + (waveM + tm * 16 + l15) * AROW_U16;
#pragma unroll
      for (int r = 0; r < 8; ++r) {           // A: VGPR r -> K pairs, half-dependent
        const int kk = hi * 16 + 2 * r;
        fa[tm].u[2 * r]     = rowp[kk];
        fa[tm].u[2 * r + 1] = rowp[kk + 1];
      }
    }
#pragma unroll
    for (int tn = 0; tn < 2; ++tn) {
      const unsigned short* colp = sB + (waveN + tn * 16 + l15) * AROW_U16;
#pragma unroll
      for (int j = 0; j < 16; ++j)            // B: lanes 0-15 K=0..15, 16-31 K=16..31
        fb[tn].u[j] = colp[hi * 16 + j];
    }
#pragma unroll
    for (int tm = 0; tm < 2; ++tm)
#pragma unroll
      for (int tn = 0; tn < 2; ++tn)
        acc[tm][tn] = __builtin_amdgcn_wmma_f32_16x16x32_bf16(
            false, fa[tm].v, false, fb[tn].v, (short)0, acc[tm][tn], false, false);
    __syncthreads();
  }

  // ---- store (C layout: VGPR r -> M = r + 8*laneHalf, N = lane&15)
  const int hi  = lane >> 4;
  const int l15 = lane & 15;
#pragma unroll
  for (int tm = 0; tm < 2; ++tm)
#pragma unroll
    for (int tn = 0; tn < 2; ++tn)
#pragma unroll
      for (int r = 0; r < 8; ++r) {
        const int gm = blockM + waveM + tm * 16 + hi * 8 + r;
        const int gn = blockN + waveN + tn * 16 + l15;
        const float val = acc[tm][tn][r];
        if (MODE == 0) {
          const int which = gn >> 9;
          const int head  = (gn >> 6) & 7;
          const int d     = gn & 63;
          float* dst = (which == 0) ? qo : (which == 1) ? ko : vo;
          dst[((size_t)head * N_SEQ + gm) * DH + d] = val;
        } else {
          out[(size_t)permOut[gm] * DIM + gn] = val + bias[gn];
        }
      }
}

// ---------------------------------------------------------------------------
// Segment-mean pooling: ql, kl [H][L][DH]
// ---------------------------------------------------------------------------
__global__ void pool_kernel(const float* __restrict__ q, const float* __restrict__ k,
                            const int* __restrict__ base,
                            float* __restrict__ ql, float* __restrict__ kl) {
  const int h = blockIdx.x >> 6;
  const int l = blockIdx.x & 63;
  const int d = threadIdx.x;                  // 64 threads
  const int s = base[l], e = base[l + 1];
  float sq = 0.f, sk = 0.f;
  for (int p = s; p < e; ++p) {
    sq += q[((size_t)h * N_SEQ + p) * DH + d];
    sk += k[((size_t)h * N_SEQ + p) * DH + d];
  }
  int cnt = e - s; if (cnt < 1) cnt = 1;
  const float inv = 1.f / (float)cnt;
  ql[((size_t)h * LSEG + l) * DH + d] = sq * inv;
  kl[((size_t)h * LSEG + l) * DH + d] = sk * inv;
}

// ---------------------------------------------------------------------------
// t3 = softmax(ql @ k^T, over n) @ v  -> [H][L][DH]   (online softmax, 2 pass)
// ---------------------------------------------------------------------------
__global__ __launch_bounds__(256)
void attn3v_kernel(const float* __restrict__ ql, const float* __restrict__ k,
                   const float* __restrict__ v, float* __restrict__ t3) {
  const int h = blockIdx.x >> 6;
  const int l = blockIdx.x & 63;
  const int t = threadIdx.x;
  __shared__ float qls[DH];
  __shared__ float red_m[256], red_s[256];
  __shared__ float wpart[8][DH];
  __shared__ float Mg, Sg;
  if (t < DH) qls[t] = ql[((size_t)h * LSEG + l) * DH + t];
  __syncthreads();
  const float* kh = k + (size_t)h * N_SEQ * DH;
  const float* vh = v + (size_t)h * N_SEQ * DH;

  float m = -1e30f, s = 0.f;
  for (int n = t; n < N_SEQ; n += 256) {
    const float* kr = kh + (size_t)n * DH;
    float dot = 0.f;
#pragma unroll 16
    for (int d = 0; d < DH; ++d) dot += qls[d] * kr[d];
    if (dot > m) { s = s * __expf(m - dot) + 1.f; m = dot; }
    else         { s += __expf(dot - m); }
  }
  red_m[t] = m; red_s[t] = s;
  __syncthreads();
  for (int off = 128; off > 0; off >>= 1) {
    if (t < off) {
      const float m1 = red_m[t], s1 = red_s[t];
      const float m2 = red_m[t + off], s2 = red_s[t + off];
      const float mm = fmaxf(m1, m2);
      red_m[t] = mm;
      red_s[t] = s1 * __expf(m1 - mm) + s2 * __expf(m2 - mm);
    }
    __syncthreads();
  }
  if (t == 0) { Mg = red_m[0]; Sg = red_s[0]; }
  __syncthreads();
  const float M = Mg, Sinv = 1.f / Sg;

  float accd[DH];
#pragma unroll
  for (int d = 0; d < DH; ++d) accd[d] = 0.f;
  for (int n = t; n < N_SEQ; n += 256) {
    const float* kr = kh + (size_t)n * DH;
    float dot = 0.f;
#pragma unroll 16
    for (int d = 0; d < DH; ++d) dot += qls[d] * kr[d];
    const float w = __expf(dot - M);
    const float* vr = vh + (size_t)n * DH;
#pragma unroll 16
    for (int d = 0; d < DH; ++d) accd[d] += w * vr[d];
  }
#pragma unroll
  for (int d = 0; d < DH; ++d) {              // wave32 reduce
    float x = accd[d];
    for (int off = 16; off > 0; off >>= 1) x += __shfl_down(x, off, 32);
    accd[d] = x;
  }
  const int lane = t & 31, wid = t >> 5;
  if (lane == 0)
    for (int d = 0; d < DH; ++d) wpart[wid][d] = accd[d];
  __syncthreads();
  if (t < DH) {
    float sum = 0.f;
#pragma unroll
    for (int w8 = 0; w8 < 8; ++w8) sum += wpart[w8][t];
    t3[((size_t)h * LSEG + l) * DH + t] = sum * Sinv;
  }
}

// ---------------------------------------------------------------------------
// attn2 = softmax(ql@kl^T); Z = pinv(attn2) (Newton-Schulz, polynomial form);
// M2 = Z @ t3.   One block per head, 3 LDS matrices.
// ---------------------------------------------------------------------------
__global__ __launch_bounds__(256)
void attn2_pinv_kernel(const float* __restrict__ ql, const float* __restrict__ kl,
                       const float* __restrict__ t3, float* __restrict__ M2) {
  const int h = blockIdx.x;
  const int t = threadIdx.x;
  __shared__ float Amat[64][64];
  __shared__ float Zmat[64][64];
  __shared__ float Tmat[64][64];
  __shared__ float red[64];
  __shared__ float cmaxS;
  const int row = t >> 2;
  const int j0  = (t & 3) * 16;
  const float* qlh = ql + (size_t)h * LSEG * DH;
  const float* klh = kl + (size_t)h * LSEG * DH;

  for (int q2 = 0; q2 < 16; ++q2) {
    const int j = j0 + q2;
    float dot = 0.f;
    for (int d = 0; d < DH; ++d) dot += qlh[row * DH + d] * klh[j * DH + d];
    Amat[row][j] = dot;
  }
  __syncthreads();
  if (t < 64) {                               // row softmax
    float mx = -1e30f;
    for (int j = 0; j < 64; ++j) mx = fmaxf(mx, Amat[t][j]);
    float s = 0.f;
    for (int j = 0; j < 64; ++j) { const float e = __expf(Amat[t][j] - mx); Amat[t][j] = e; s += e; }
    const float inv = 1.f / s;
    for (int j = 0; j < 64; ++j) Amat[t][j] *= inv;
  }
  __syncthreads();
  if (t < 64) {                               // column sums (row sums == 1)
    float cs = 0.f;
    for (int i = 0; i < 64; ++i) cs += Amat[i][t];
    red[t] = cs;
  }
  __syncthreads();
  if (t == 0) {
    float mx = red[0];
    for (int j = 1; j < 64; ++j) mx = fmaxf(mx, red[j]);
    cmaxS = mx;
  }
  __syncthreads();
  const float zscale = 1.f / cmaxS;
  for (int q2 = 0; q2 < 16; ++q2) Zmat[row][j0 + q2] = Amat[j0 + q2][row] * zscale;
  __syncthreads();

  float r[16], zacc[16];
  for (int it = 0; it < 6; ++it) {
    // T = A @ Z
    for (int q2 = 0; q2 < 16; ++q2) {
      float a = 0.f;
      for (int kk = 0; kk < 64; ++kk) a += Amat[row][kk] * Zmat[kk][j0 + q2];
      r[q2] = a;
    }
    __syncthreads();
    for (int q2 = 0; q2 < 16; ++q2) Tmat[row][j0 + q2] = r[q2];
    __syncthreads();
    // Z <- 0.25*(13Z - 15 Z T + 7 Z T^2 - Z T^3)   (powers of T=AZ commute)
    for (int q2 = 0; q2 < 16; ++q2) {
      float a = 0.f;
      for (int kk = 0; kk < 64; ++kk) a += Zmat[row][kk] * Tmat[kk][j0 + q2];
      r[q2] = a;
      zacc[q2] = 13.f * Zmat[row][j0 + q2] - 15.f * a;
    }
    __syncthreads();
    for (int q2 = 0; q2 < 16; ++q2) Zmat[row][j0 + q2] = r[q2];       // Z = P1
    __syncthreads();
    for (int q2 = 0; q2 < 16; ++q2) {
      float a = 0.f;
      for (int kk = 0; kk < 64; ++kk) a += Zmat[row][kk] * Tmat[kk][j0 + q2];
      r[q2] = a; zacc[q2] += 7.f * a;
    }
    __syncthreads();
    for (int q2 = 0; q2 < 16; ++q2) Zmat[row][j0 + q2] = r[q2];       // Z = P2
    __syncthreads();
    for (int q2 = 0; q2 < 16; ++q2) {
      float a = 0.f;
      for (int kk = 0; kk < 64; ++kk) a += Zmat[row][kk] * Tmat[kk][j0 + q2];
      zacc[q2] -= a;
    }
    __syncthreads();
    for (int q2 = 0; q2 < 16; ++q2) Zmat[row][j0 + q2] = 0.25f * zacc[q2];
    __syncthreads();
  }
  // M2 = Z @ t3[h]
  const float* t3h = t3 + (size_t)h * LSEG * DH;
  for (int q2 = 0; q2 < 16; ++q2) Tmat[row][j0 + q2] = t3h[row * DH + j0 + q2];
  __syncthreads();
  for (int q2 = 0; q2 < 16; ++q2) {
    float a = 0.f;
    for (int kk = 0; kk < 64; ++kk) a += Zmat[row][kk] * Tmat[kk][j0 + q2];
    M2[((size_t)h * LSEG + row) * DH + j0 + q2] = a;
  }
}

// ---------------------------------------------------------------------------
// out_inner[p][h*64+d] = softmax_l(q[h,p]·kl[h,l]) @ M2[h]     (fp32)
// ---------------------------------------------------------------------------
__global__ __launch_bounds__(256)
void attn1_kernel(const float* __restrict__ q, const float* __restrict__ kl,
                  const float* __restrict__ M2, float* __restrict__ out_inner) {
  const int h = blockIdx.y;
  const int p = blockIdx.x * 256 + threadIdx.x;
  const int t = threadIdx.x;
  __shared__ float kls[64][64];
  __shared__ float m2s[64][64];
  for (int e = t; e < 4096; e += 256) {
    kls[e >> 6][e & 63] = kl[(size_t)h * LSEG * DH + e];
    m2s[e >> 6][e & 63] = M2[(size_t)h * LSEG * DH + e];
  }
  __syncthreads();
  float qv[64];
  const float* qr = q + ((size_t)h * N_SEQ + p) * DH;
#pragma unroll
  for (int d = 0; d < 64; ++d) qv[d] = qr[d];
  float w[64];
  float mx = -1e30f;
  for (int l = 0; l < 64; ++l) {
    float dot = 0.f;
#pragma unroll 16
    for (int d = 0; d < 64; ++d) dot += qv[d] * kls[l][d];
    w[l] = dot; mx = fmaxf(mx, dot);
  }
  float s = 0.f;
  for (int l = 0; l < 64; ++l) { w[l] = __expf(w[l] - mx); s += w[l]; }
  const float inv = 1.f / s;
  float* op = out_inner + (size_t)p * INNER + h * DH;
  for (int d = 0; d < 64; ++d) {
    float o = 0.f;
#pragma unroll 16
    for (int l = 0; l < 64; ++l) o += w[l] * m2s[l][d];
    op[d] = o * inv;
  }
}

// ---------------------------------------------------------------------------
// Depthwise conv along sorted n (KS=33) + attn1 result -> bf16 inner matrix
// ---------------------------------------------------------------------------
__global__ __launch_bounds__(256)
void conv_store_bf16(const float* __restrict__ v, const float* __restrict__ rk,
                     const float* __restrict__ out_inner,
                     unsigned short* __restrict__ inner_bf) {
  const size_t tid = (size_t)blockIdx.x * 256 + threadIdx.x;   // H*N*DH = 2^24
  const int d = (int)(tid & 63);
  const int p = (int)((tid >> 6) & (N_SEQ - 1));
  const int h = (int)(tid >> 21);
  const float* vh = v + (size_t)h * N_SEQ * DH;
  float acc = 0.f;
#pragma unroll
  for (int j = 0; j < KS; ++j) {
    const int pp = p + j - KS / 2;
    if (pp >= 0 && pp < N_SEQ) acc += rk[h * KS + j] * vh[(size_t)pp * DH + d];
  }
  const size_t o = (size_t)p * INNER + h * DH + d;
  inner_bf[o] = f2bf(out_inner[o] + acc);
}

// ---------------------------------------------------------------------------
extern "C" void kernel_launch(void* const* d_in, const int* in_sizes, int n_in,
                              void* d_out, int out_size, void* d_ws, size_t ws_size,
                              hipStream_t stream) {
  const float* x      = (const float*)d_in[0];
  const float* Wqkv   = (const float*)d_in[1];
  const float* Wout   = (const float*)d_in[2];
  const float* bout   = (const float*)d_in[3];
  const float* rk     = (const float*)d_in[4];
  const int*   labels = (const int*)d_in[5];
  float* out = (float*)d_out;

  char* ws = (char*)d_ws;
  size_t off = 0;
  auto alloc = [&](size_t bytes) -> void* {
    void* p = ws + off;
    off += (bytes + 255) & ~(size_t)255;
    return p;
  };

  int*   perm      = (int*)alloc(sizeof(int) * N_SEQ);
  int*   blockHist = (int*)alloc(sizeof(int) * SORT_BLOCKS * 64);
  int*   blockOff  = (int*)alloc(sizeof(int) * SORT_BLOCKS * 64);
  int*   base      = (int*)alloc(sizeof(int) * 65);
  unsigned short* xbf     = (unsigned short*)alloc(sizeof(unsigned short) * (size_t)N_SEQ * DIM);
  unsigned short* wqkvT   = (unsigned short*)alloc(sizeof(unsigned short) * (size_t)DIM * THREE_INNER);
  unsigned short* woutT   = (unsigned short*)alloc(sizeof(unsigned short) * (size_t)INNER * DIM);
  unsigned short* innerBf = (unsigned short*)alloc(sizeof(unsigned short) * (size_t)N_SEQ * INNER);
  float* q  = (float*)alloc(sizeof(float) * (size_t)H * N_SEQ * DH);
  float* k  = (float*)alloc(sizeof(float) * (size_t)H * N_SEQ * DH);
  float* v  = (float*)alloc(sizeof(float) * (size_t)H * N_SEQ * DH);
  float* ql = (float*)alloc(sizeof(float) * H * LSEG * DH);
  float* kl = (float*)alloc(sizeof(float) * H * LSEG * DH);
  float* t3 = (float*)alloc(sizeof(float) * H * LSEG * DH);
  float* M2 = (float*)alloc(sizeof(float) * H * LSEG * DH);
  float* out_inner = (float*)alloc(sizeof(float) * (size_t)N_SEQ * INNER);

  hist_kernel<<<SORT_BLOCKS, 256, 0, stream>>>(labels, blockHist);
  scan_kernel<<<1, 64, 0, stream>>>(blockHist, blockOff, base);
  scatter_kernel<<<SORT_BLOCKS, 256, 0, stream>>>(labels, blockOff, perm);

  transpose_bf16<<<(DIM * THREE_INNER + 255) / 256, 256, 0, stream>>>(Wqkv, wqkvT, DIM, THREE_INNER);
  transpose_bf16<<<(INNER * DIM + 255) / 256, 256, 0, stream>>>(Wout, woutT, INNER, DIM);
  gather_x_bf16<<<(N_SEQ * DIM / 4) / 256, 256, 0, stream>>>(x, perm, xbf);

  gemm_bf16_wmma<0><<<dim3(THREE_INNER / 64, N_SEQ / 128), 256, GEMM_SMEM_BYTES, stream>>>(
      xbf, wqkvT, DIM, THREE_INNER, q, k, v, nullptr, nullptr, nullptr);

  pool_kernel<<<H * LSEG, 64, 0, stream>>>(q, k, base, ql, kl);
  attn3v_kernel<<<H * LSEG, 256, 0, stream>>>(ql, k, v, t3);
  attn2_pinv_kernel<<<H, 256, 0, stream>>>(ql, kl, t3, M2);
  attn1_kernel<<<dim3(N_SEQ / 256, H), 256, 0, stream>>>(q, kl, M2, out_inner);
  conv_store_bf16<<<(H * N_SEQ * DH) / 256, 256, 0, stream>>>(v, rk, out_inner, innerBf);

  gemm_bf16_wmma<1><<<dim3(DIM / 64, N_SEQ / 128), 256, GEMM_SMEM_BYTES, stream>>>(
      innerBf, woutT, INNER, DIM, nullptr, nullptr, nullptr, perm, bout, out);
}
